// TransformerEncoderLayer_81114752352882
// MI455X (gfx1250) — matrix operand order repaired and checked by reading, not compiled, not run on hardware
//
#include <hip/hip_runtime.h>
#include <hip/hip_bf16.h>

// ---------------------------------------------------------------------------
// Fixed problem shape (from the reference)
// ---------------------------------------------------------------------------
constexpr int B_  = 2;
constexpr int T_  = 2048;
constexpr int D_  = 1024;
constexpr int H_  = 16;
constexpr int HD_ = 64;       // D / H
constexpr int FF_ = 4096;
constexpr int NT_ = B_ * T_;  // 4096 total rows

typedef __attribute__((ext_vector_type(16))) _Float16 v16h;
typedef __attribute__((ext_vector_type(8)))  float    v8f;

// ---------------------------------------------------------------------------
// WMMA fragment loaders (16x16x32 f16, wave32; contiguous k order used here, not the ISA table order)
//
// A (16x32, MxK): lane L holds row M = L&15.
//   lanes 0-15 : elems 0..15 -> K=kb+0..15
//   lanes 16-31: elems 0..15 -> K=kb+16..31
// B (32x16, KxN): lane L holds col N = L&15.
//   lanes 0-15 : elems 0..15 -> K=kb+0..15
//   lanes 16-31: elems 0..15 -> K=kb+16..31
// Both are contiguous 16B chunks when the operand is stored K-innermost.
// ---------------------------------------------------------------------------
__device__ __forceinline__ v16h load_a_frag(const _Float16* __restrict__ rowp, int lane) {
  const _Float16* p = rowp + ((lane >> 4) << 4);       // +16 halves for upper half-wave
  v16h f;
  reinterpret_cast<uint4*>(&f)[0] = *reinterpret_cast<const uint4*>(p);
  reinterpret_cast<uint4*>(&f)[1] = *reinterpret_cast<const uint4*>(p + 8);
  return f;
}

// Works for both global and LDS pointers (16 contiguous halves per lane-half).
__device__ __forceinline__ v16h load_b_frag(const _Float16* colp, int lane) {
  const _Float16* p = colp + ((lane >> 4) << 4);       // +16 halves for upper half-wave
  v16h f;
  reinterpret_cast<uint4*>(&f)[0] = *reinterpret_cast<const uint4*>(p);
  reinterpret_cast<uint4*>(&f)[1] = *reinterpret_cast<const uint4*>(p + 8);
  return f;
}

// CDNA5 async copy: 16 bytes global -> LDS per lane, tracked by ASYNCcnt.
__device__ __forceinline__ void async_b128_to_lds(const void* smem_dst, const void* gsrc) {
  const uint32_t l = (uint32_t)(uintptr_t)smem_dst;     // low 32 bits = LDS offset
  const uint64_t g = (uint64_t)(uintptr_t)gsrc;
  asm volatile("global_load_async_to_lds_b128 %0, %1, off" :: "v"(l), "v"(g) : "memory");
}

// ---------------------------------------------------------------------------
// WMMA GEMM:  C[M,N] = A[M,K] (f16, lda) x Bt[N,K] (f16, ldb)
// 256 threads = 8 waves arranged 4(M) x 2(N).
// Wave tile: (MT*16) x (NT*16).  Block tile: (4*MT*16) x (2*NT*16).
//
// The B tile for the whole block is staged through LDS with
// global_load_async_to_lds_b128 (double-buffered, 64-half K chunks) so the 4
// M-wave rows share one copy of B; A is read direct from global with a
// register ping-pong.  s_wait_asynccnt NT leaves the *next* chunk's copies in
// flight while the current chunk's WMMAs execute (ASYNCcnt decrements in
// order).  Grids divide shapes exactly -> EXEC all-ones at every v_wmma.
// MODE 0: f16 out = acc*scale + bias
// MODE 1: f16 out = relu(acc*scale + bias)
// MODE 2: f16 out stored transposed per-head: vT[B][H][HD][T]
// MODE 3: f32 out = acc*scale                  (attention scores)
// MODE 4: f32 out = acc*scale + bias + resid   (residual adds)
// ---------------------------------------------------------------------------
template <int MT, int NT, int MODE>
__global__ __launch_bounds__(256) void k_gemm(
    const _Float16* __restrict__ A,  int lda,
    const _Float16* __restrict__ Bt, int ldb,
    const float* __restrict__ bias,
    const float* __restrict__ resid,
    float scale,
    void* __restrict__ outv, int ldc,
    int K)
{
  constexpr int ROWS = 2 * NT * 16;                     // B rows per block tile
  __shared__ alignas(16) _Float16 smem[2][ROWS * 64];   // [buf][row][64 halves]

  const int lane   = threadIdx.x & 31;
  const int wave   = threadIdx.x >> 5;
  const int mw     = wave & 3;
  const int nw     = wave >> 2;
  const int blockN = blockIdx.x * ROWS;
  const int waveM  = blockIdx.y * (4 * MT * 16) + mw * (MT * 16);

  const _Float16* aptr[MT];
#pragma unroll
  for (int mt = 0; mt < MT; ++mt)
    aptr[mt] = A + (size_t)(waveM + mt * 16 + (lane & 15)) * lda;

  // LDS rows this wave's B fragments come from (local to the block tile).
  int brow[NT];
#pragma unroll
  for (int t = 0; t < NT; ++t)
    brow[t] = nw * (NT * 16) + t * 16 + (lane & 15);

  v8f acc[MT][NT] = {};
  v16h a0[MT], a1[MT], b0[NT], b1[NT];

  // Stage one 64-half K chunk of the block's B tile into LDS buffer `buf`.
  // ROWS*8 b128 transfers total -> NT per thread; ASYNCcnt += NT per wave.
  auto stageB = [&](int buf, int kb) {
#pragma unroll
    for (int j = 0; j < NT; ++j) {
      const int idx = j * 256 + (int)threadIdx.x;
      const int row = idx >> 3;
      const int seg = idx & 7;
      async_b128_to_lds(&smem[buf][row * 64 + seg * 8],
                        Bt + (size_t)(blockN + row) * ldb + kb + seg * 8);
    }
  };
  auto loadA = [&](v16h* a, int kb) {
#pragma unroll
    for (int mt = 0; mt < MT; ++mt) a[mt] = load_a_frag(aptr[mt] + kb, lane);
  };
  auto mma = [&](const v16h* a, const v16h* b) {
#pragma unroll
    for (int mt = 0; mt < MT; ++mt)
#pragma unroll
      for (int t = 0; t < NT; ++t)
        acc[mt][t] = __builtin_amdgcn_wmma_f32_16x16x32_f16(
            false, a[mt], false, b[t], (short)0, acc[mt][t], false, false);
  };

  // K is always a multiple of 64 here (64 / 1024 / 2048 / 4096).
  stageB(0, 0);
  for (int kb = 0; kb < K; kb += 64) {
    const int  cur     = (kb >> 6) & 1;
    const bool hasNext = (kb + 64) < K;
    if (hasNext) stageB(cur ^ 1, kb + 64);
    loadA(a0, kb);
    loadA(a1, kb + 32);
    if (hasNext) {
      if (NT == 4) asm volatile("s_wait_asynccnt 0x4" ::: "memory");
      else         asm volatile("s_wait_asynccnt 0x2" ::: "memory");
    } else {
      asm volatile("s_wait_asynccnt 0x0" ::: "memory");
    }
    __syncthreads();                    // chunk `cur` visible to all waves
    const _Float16* sb = smem[cur];
#pragma unroll
    for (int t = 0; t < NT; ++t) b0[t] = load_b_frag(sb + brow[t] * 64, lane);
    mma(a0, b0);
#pragma unroll
    for (int t = 0; t < NT; ++t) b1[t] = load_b_frag(sb + brow[t] * 64 + 32, lane);
    mma(a1, b1);
    __syncthreads();                    // all reads done before buf is re-staged
  }

  // C/D layout: lane&15 = N, VGPR v = M offset (upper half-wave += 8)
#pragma unroll
  for (int mt = 0; mt < MT; ++mt) {
    const int rbase = waveM + mt * 16 + ((lane >> 4) << 3);
#pragma unroll
    for (int t = 0; t < NT; ++t) {
      const int col = blockN + nw * (NT * 16) + t * 16 + (lane & 15);
      const float bv = bias ? bias[col] : 0.0f;
#pragma unroll
      for (int v = 0; v < 8; ++v) {
        const int row = rbase + v;
        float val = acc[mt][t][v] * scale + bv;
        if (MODE == 1) val = fmaxf(val, 0.0f);
        if (MODE == 0 || MODE == 1) {
          reinterpret_cast<_Float16*>(outv)[(size_t)row * ldc + col] = (_Float16)val;
        } else if (MODE == 2) {
          const int bb = row >> 11;          // row / T_
          const int tt = row & (T_ - 1);
          const int hh = col >> 6;           // col / HD_
          const int dd = col & (HD_ - 1);
          reinterpret_cast<_Float16*>(outv)[((size_t)((bb * H_ + hh) * HD_ + dd)) * T_ + tt] =
              (_Float16)val;
        } else if (MODE == 3) {
          reinterpret_cast<float*>(outv)[(size_t)row * ldc + col] = val;
        } else {  // MODE == 4
          reinterpret_cast<float*>(outv)[(size_t)row * ldc + col] =
              val + resid[(size_t)row * ldc + col];
        }
      }
    }
  }
}

// ---------------------------------------------------------------------------
// Weight convert + transpose: wT[n*K + k] = (f16) w[k*N + n]
// ---------------------------------------------------------------------------
__global__ __launch_bounds__(256) void k_cvtT(const float* __restrict__ w,
                                              _Float16* __restrict__ wT,
                                              int K, int N)
{
  const size_t idx = (size_t)blockIdx.x * 256 + threadIdx.x;
  if (idx >= (size_t)K * N) return;
  const int n = (int)(idx / K);
  const int k = (int)(idx % K);
  wT[idx] = (_Float16)w[(size_t)k * N + n];
}

// ---------------------------------------------------------------------------
// LayerNorm over D_ (one block per row), f32 in -> f16 out
// ---------------------------------------------------------------------------
__global__ __launch_bounds__(256) void k_layernorm(const float* __restrict__ x,
                                                   const float* __restrict__ g,
                                                   const float* __restrict__ b,
                                                   _Float16* __restrict__ out)
{
  __shared__ float sdata[256];
  const int row = blockIdx.x;
  const float* xr = x + (size_t)row * D_;

  float s = 0.0f;
  for (int i = threadIdx.x; i < D_; i += 256) s += xr[i];
  sdata[threadIdx.x] = s; __syncthreads();
  for (int off = 128; off > 0; off >>= 1) {
    if (threadIdx.x < off) sdata[threadIdx.x] += sdata[threadIdx.x + off];
    __syncthreads();
  }
  const float mu = sdata[0] * (1.0f / D_);
  __syncthreads();

  float v = 0.0f;
  for (int i = threadIdx.x; i < D_; i += 256) { const float d = xr[i] - mu; v += d * d; }
  sdata[threadIdx.x] = v; __syncthreads();
  for (int off = 128; off > 0; off >>= 1) {
    if (threadIdx.x < off) sdata[threadIdx.x] += sdata[threadIdx.x + off];
    __syncthreads();
  }
  const float rstd = rsqrtf(sdata[0] * (1.0f / D_) + 1e-5f);
  __syncthreads();

  _Float16* orow = out + (size_t)row * D_;
  for (int i = threadIdx.x; i < D_; i += 256)
    orow[i] = (_Float16)((xr[i] - mu) * rstd * g[i] + b[i]);
}

// ---------------------------------------------------------------------------
// Row softmax over T_ keys; writes f16 probs and accumulates probs/H into the
// avg_attn output slab (heads are processed sequentially -> plain RMW is safe)
// ---------------------------------------------------------------------------
__global__ __launch_bounds__(256) void k_softmax(const float* __restrict__ scores,
                                                 _Float16* __restrict__ probs,
                                                 float* __restrict__ avg)
{
  __shared__ float sdata[256];
  const int row = blockIdx.x;
  const float* sr = scores + (size_t)row * T_;

  float m = -3.4e38f;
  for (int i = threadIdx.x; i < T_; i += 256) m = fmaxf(m, sr[i]);
  sdata[threadIdx.x] = m; __syncthreads();
  for (int off = 128; off > 0; off >>= 1) {
    if (threadIdx.x < off) sdata[threadIdx.x] = fmaxf(sdata[threadIdx.x], sdata[threadIdx.x + off]);
    __syncthreads();
  }
  m = sdata[0]; __syncthreads();

  float s = 0.0f;
  for (int i = threadIdx.x; i < T_; i += 256) s += __expf(sr[i] - m);
  sdata[threadIdx.x] = s; __syncthreads();
  for (int off = 128; off > 0; off >>= 1) {
    if (threadIdx.x < off) sdata[threadIdx.x] += sdata[threadIdx.x + off];
    __syncthreads();
  }
  const float inv = 1.0f / sdata[0];

  _Float16* pr = probs + (size_t)row * T_;
  float*    ar = avg   + (size_t)row * T_;
  for (int i = threadIdx.x; i < T_; i += 256) {
    const float p = __expf(sr[i] - m) * inv;
    pr[i] = (_Float16)p;
    ar[i] += p * (1.0f / H_);
  }
}

__global__ __launch_bounds__(256) void k_zero(float* __restrict__ p, size_t n)
{
  const size_t i = (size_t)blockIdx.x * 256 + threadIdx.x;
  if (i < n) p[i] = 0.0f;
}

// ---------------------------------------------------------------------------
// Host-side orchestration (all launches on `stream`; graph-capture safe)
// ---------------------------------------------------------------------------
extern "C" void kernel_launch(void* const* d_in, const int* in_sizes, int n_in,
                              void* d_out, int out_size, void* d_ws, size_t ws_size,
                              hipStream_t stream) {
  (void)in_sizes; (void)n_in; (void)out_size; (void)ws_size;

  const float* x     = (const float*)d_in[0];
  const float* ln1_g = (const float*)d_in[1];
  const float* ln1_b = (const float*)d_in[2];
  const float* wq    = (const float*)d_in[3];
  const float* bq    = (const float*)d_in[4];
  const float* wk    = (const float*)d_in[5];
  const float* bk    = (const float*)d_in[6];
  const float* wv    = (const float*)d_in[7];
  const float* bv    = (const float*)d_in[8];
  const float* wo    = (const float*)d_in[9];
  const float* bo    = (const float*)d_in[10];
  const float* ln2_g = (const float*)d_in[11];
  const float* ln2_b = (const float*)d_in[12];
  const float* w1    = (const float*)d_in[13];
  const float* b1    = (const float*)d_in[14];
  const float* w2    = (const float*)d_in[15];
  const float* b2    = (const float*)d_in[16];

  float* out_x   = (float*)d_out;                      // [B,T,D]
  float* out_avg = out_x + (size_t)NT_ * D_;           // [B,T,T]

  // ---- workspace carve-out (~144 MB) ----
  char*  ws  = (char*)d_ws;
  size_t off = 0;
  auto alloc = [&](size_t bytes) -> void* {
    void* p = ws + off;
    off = (off + bytes + 255) & ~(size_t)255;
    return p;
  };
  _Float16* h16    = (_Float16*)alloc((size_t)NT_ * D_  * 2);
  _Float16* wqT    = (_Float16*)alloc((size_t)D_  * D_  * 2);
  _Float16* wkT    = (_Float16*)alloc((size_t)D_  * D_  * 2);
  _Float16* wvT    = (_Float16*)alloc((size_t)D_  * D_  * 2);
  _Float16* woT    = (_Float16*)alloc((size_t)D_  * D_  * 2);
  _Float16* w1T    = (_Float16*)alloc((size_t)D_  * FF_ * 2);
  _Float16* w2T    = (_Float16*)alloc((size_t)FF_ * D_  * 2);
  _Float16* q16    = (_Float16*)alloc((size_t)NT_ * D_  * 2);   // [B,T,H,HD]
  _Float16* k16    = (_Float16*)alloc((size_t)NT_ * D_  * 2);   // [B,T,H,HD]
  _Float16* vT16   = (_Float16*)alloc((size_t)NT_ * D_  * 2);   // [B,H,HD,T]
  _Float16* attn16 = (_Float16*)alloc((size_t)NT_ * D_  * 2);   // [B,T,D]
  float*    x1     = (float*)   alloc((size_t)NT_ * D_  * 4);   // x + attn_out
  _Float16* h2_16  = (_Float16*)alloc((size_t)NT_ * D_  * 2);
  _Float16* ff16   = (_Float16*)alloc((size_t)NT_ * FF_ * 2);
  float*    scores = (float*)   alloc((size_t)T_  * T_  * 4);   // reused per (b,h)
  _Float16* probs  = (_Float16*)alloc((size_t)T_  * T_  * 2);   // reused per (b,h)

  // ---- 0: zero the avg_attn output slab (it is accumulated into) ----
  {
    const size_t n = (size_t)B_ * T_ * T_;
    k_zero<<<dim3((unsigned)((n + 255) / 256)), 256, 0, stream>>>(out_avg, n);
  }

  // ---- 1: convert + transpose weights to f16 [N,K] ----
  auto cvt = [&](const float* w, _Float16* wT, int K, int N) {
    const size_t tot = (size_t)K * N;
    k_cvtT<<<dim3((unsigned)((tot + 255) / 256)), 256, 0, stream>>>(w, wT, K, N);
  };
  cvt(wq, wqT, D_, D_);
  cvt(wk, wkT, D_, D_);
  cvt(wv, wvT, D_, D_);
  cvt(wo, woT, D_, D_);
  cvt(w1, w1T, D_, FF_);
  cvt(w2, w2T, FF_, D_);

  // ---- 2: LN1 ----
  k_layernorm<<<NT_, 256, 0, stream>>>(x, ln1_g, ln1_b, h16);

  // ---- 3: Q/K/V projections (M=4096, N=1024, K=1024); block tile 128x128 ----
  const dim3 gProj(D_ / 128, NT_ / 128);
  k_gemm<2, 4, 0><<<gProj, 256, 0, stream>>>(h16, D_, wqT, D_, bq, nullptr, 1.0f, q16, D_, D_);
  k_gemm<2, 4, 0><<<gProj, 256, 0, stream>>>(h16, D_, wkT, D_, bk, nullptr, 1.0f, k16, D_, D_);
  k_gemm<2, 4, 2><<<gProj, 256, 0, stream>>>(h16, D_, wvT, D_, bv, nullptr, 1.0f, vT16, 0, D_);

  // ---- 4: attention, one (b,h) at a time (bounds workspace to 24 MB) ----
  for (int b = 0; b < B_; ++b) {
    float* avg_b = out_avg + (size_t)b * T_ * T_;
    for (int h = 0; h < H_; ++h) {
      const _Float16* qh = q16 + (size_t)(b * T_) * D_ + h * HD_;   // A: [T, 64], lda=D
      const _Float16* kh = k16 + (size_t)(b * T_) * D_ + h * HD_;   // Bt: [T, 64], ldb=D
      // scores = q k^T / sqrt(HD)   (M=N=2048, K=64)
      k_gemm<2, 4, 3><<<dim3(T_ / 128, T_ / 128), 256, 0, stream>>>(
          qh, D_, kh, D_, nullptr, nullptr, 0.125f, scores, T_, HD_);
      // softmax rows + accumulate probs/H into avg_attn
      k_softmax<<<T_, 256, 0, stream>>>(scores, probs, avg_b);
      // attn = probs @ v   (M=2048, N=64, K=2048); Bt = vT[b][h] : [64, T]
      const _Float16* vh = vT16 + (size_t)((b * H_ + h) * HD_) * T_;
      _Float16*       ah = attn16 + (size_t)(b * T_) * D_ + h * HD_;
      k_gemm<2, 2, 0><<<dim3(1, T_ / 128), 256, 0, stream>>>(
          probs, T_, vh, T_, nullptr, nullptr, 1.0f, ah, D_, T_);
    }
  }

  // ---- 5: out-projection + residual: x1 = x + attn @ wo + bo ----
  k_gemm<2, 4, 4><<<gProj, 256, 0, stream>>>(attn16, D_, woT, D_, bo, x, 1.0f, x1, D_, D_);

  // ---- 6: LN2 ----
  k_layernorm<<<NT_, 256, 0, stream>>>(x1, ln2_g, ln2_b, h2_16);

  // ---- 7: FFN1 with ReLU (M=4096, N=4096, K=1024) ----
  k_gemm<2, 4, 1><<<dim3(FF_ / 128, NT_ / 128), 256, 0, stream>>>(
      h2_16, D_, w1T, D_, b1, nullptr, 1.0f, ff16, FF_, D_);

  // ---- 8: FFN2 + residual -> final output (M=4096, N=1024, K=4096) ----
  k_gemm<2, 4, 4><<<gProj, 256, 0, stream>>>(
      ff16, FF_, w2T, FF_, b2, x1, 1.0f, out_x, D_, FF_);
}
